// AttentionBlock_32469952757987
// MI455X (gfx1250) — matrix operand order repaired and checked by reading, not compiled, not run on hardware
//
#include <hip/hip_runtime.h>

typedef __bf16 bf16;
typedef __attribute__((ext_vector_type(16))) __bf16 v16bf;
typedef __attribute__((ext_vector_type(8)))  __bf16 v8bf;
typedef __attribute__((ext_vector_type(4)))  __bf16 v4bf;
typedef __attribute__((ext_vector_type(8)))  float  v8f;
typedef __attribute__((ext_vector_type(4)))  unsigned int v4u;
typedef __attribute__((ext_vector_type(8)))  int v8i;
typedef __attribute__((ext_vector_type(4)))  int v4i;

// Problem constants (B=4, L=S=1024, DIM=1024, NH=16, DH=64)
constexpr int BB   = 4;
constexpr int LL   = 1024;
constexpr int SS   = 1024;
constexpr int DIMC = 1024;
constexpr int NHC  = 16;
constexpr int DHC  = 64;

union AFrag { v16bf v; v8bf h[2]; bf16 e[16]; };

// gfx1250 has native bf16 converts (v_cvt_pk_bf16_f32) — let clang lower it.
__device__ __forceinline__ bf16 f2bf(float f) { return (bf16)f; }

// ---------------------------------------------------------------------------
// Tensor Data Mover: 2-D bf16 tile (TW contiguous elems x TH rows, row stride
// STRIDE elems) from global -> LDS.  D# per CDNA5 ISA §8 (group0/group1).
// This toolchain exposes the 6-arg builtin: (g0, g1, g2, g3, g4, cpol).
// ---------------------------------------------------------------------------
#if __has_builtin(__builtin_amdgcn_tensor_load_to_lds) && \
    __has_builtin(__builtin_amdgcn_s_wait_tensorcnt)
#define HAS_TDM 1
#else
#define HAS_TDM 0
#endif

template <int TW, int TH, long long STRIDE>
__device__ __forceinline__ void tdm_load_tile(bf16* lds_dst, const bf16* g) {
#if HAS_TDM
    const unsigned long long ga = (unsigned long long)(uintptr_t)g;
    const unsigned ldsoff = (unsigned)(uintptr_t)lds_dst;   // low 32 bits = LDS offset
    v4u g0;
    g0[0] = 1u;                                             // count=1, user mode
    g0[1] = ldsoff;                                         // lds_addr
    g0[2] = (unsigned)ga;                                   // global_addr[31:0]
    g0[3] = (unsigned)((ga >> 32) & 0x01FFFFFFu) | (2u << 30); // addr[56:32] | type=2
    v8i g1;
    g1[0] = (int)(1u << 16);                                // data_size=1 (2 bytes)
    g1[1] = (int)(((unsigned)TW & 0xFFFFu) << 16);          // tensor_dim0 lo
    g1[2] = (int)(((unsigned)TH & 0xFFFFu) << 16);          // tensor_dim0 hi=0 | dim1 lo
    g1[3] = (int)(((unsigned)TW) << 16);                    // dim1 hi=0 | tile_dim0
    g1[4] = (int)((unsigned)TH & 0xFFFFu);                  // tile_dim1 | tile_dim2=0
    g1[5] = (int)(unsigned)(STRIDE & 0xFFFFFFFFll);         // tensor_dim0_stride lo
    g1[6] = (int)(unsigned)((STRIDE >> 32) & 0xFFFFll);     // stride hi | dim1_stride=0
    g1[7] = 0;
    v4i z4 = {};
    v8i z8 = {};
    __builtin_amdgcn_tensor_load_to_lds(g0, g1, z4, z4, z8, 0);
#else
    const int lane = threadIdx.x & 31;                      // fallback: wave copies
    for (int i = lane; i < TW * TH; i += 32)
        lds_dst[(i / TW) * TW + (i % TW)] = g[(long long)(i / TW) * STRIDE + (i % TW)];
#endif
}
__device__ __forceinline__ void tdm_wait0() {
#if HAS_TDM
    __builtin_amdgcn_s_wait_tensorcnt(0);
#endif
}

// ---------------------------------------------------------------------------
// Elementwise fp32 -> bf16 (same layout), vectorized x4
// ---------------------------------------------------------------------------
__global__ __launch_bounds__(256) void cvt_f32_bf16(const float* __restrict__ in,
                                                    bf16* __restrict__ out) {
    const int i = (blockIdx.x * 256 + threadIdx.x) * 4;
    const float4 f = *(const float4*)(in + i);
    v4bf o4;
    o4[0] = f2bf(f.x); o4[1] = f2bf(f.y); o4[2] = f2bf(f.z); o4[3] = f2bf(f.w);
    *(v4bf*)(out + i) = o4;
}

// ---------------------------------------------------------------------------
// W (K x N fp32) -> Wt (N x K bf16), LDS-tiled 32x32 transpose
// ---------------------------------------------------------------------------
__global__ __launch_bounds__(256) void transpose_w_bf16(const float* __restrict__ W,
                                                        bf16* __restrict__ Wt) {
    __shared__ float t[32][33];
    const int bx = blockIdx.x * 32, by = blockIdx.y * 32;
    const int tx = threadIdx.x & 31, ty4 = (threadIdx.x >> 5) * 4;
#pragma unroll
    for (int r = 0; r < 4; ++r)
        t[ty4 + r][tx] = W[(size_t)(by + ty4 + r) * DIMC + bx + tx];
    __syncthreads();
#pragma unroll
    for (int r = 0; r < 4; ++r)
        Wt[(size_t)(bx + ty4 + r) * DIMC + by + tx] = f2bf(t[tx][ty4 + r]);
}

// ---------------------------------------------------------------------------
// GEMM: C[4096,1024] = A_bf16(M,K) * Wt_bf16(N,K)^T + bias.  TDM double-buffered
// tile staging, 128x128 block tile, 8 waves x (16-row strip x 128 cols),
// bf16 WMMA 16x16x32 (8 per wave per 32-K step).
// MODE 0: bf16 out head layout; MODE 1: bf16 out V-transposed; MODE 2: fp32 out.
// ---------------------------------------------------------------------------
template <int MODE>
__global__ __launch_bounds__(256) void gemm_tdm_wmma(
    const bf16* __restrict__ A, const bf16* __restrict__ Wt,
    const float* __restrict__ bias, bf16* __restrict__ outb,
    float* __restrict__ outf)
{
    __shared__ __attribute__((aligned(16))) bf16 As[2][128][32];   // 16 KB
    __shared__ __attribute__((aligned(16))) bf16 Bs[2][128][32];   // 16 KB

    const int tid = threadIdx.x;
    const int wv = tid >> 5, lane = tid & 31, grp = lane >> 4, lm = lane & 15;
    const int tM = blockIdx.y * 128, tN = blockIdx.x * 128;

    v8f acc[8] = {};

    if (wv == 0) {                                      // prologue: stage k0=0
        tdm_load_tile<32, 128, DIMC>(&As[0][0][0], A + (size_t)tM * DIMC);
        tdm_load_tile<32, 128, DIMC>(&Bs[0][0][0], Wt + (size_t)tN * DIMC);
    }

    for (int k0 = 0; k0 < DIMC; k0 += 32) {
        const int buf = (k0 >> 5) & 1;
        if (wv == 0) tdm_wait0();                       // tile k0 arrived
        __syncthreads();                                // publish; prev buf free
        if (wv == 0 && k0 + 32 < DIMC) {                // overlap next tile DMA
            tdm_load_tile<32, 128, DIMC>(&As[buf ^ 1][0][0],
                                         A + (size_t)tM * DIMC + k0 + 32);
            tdm_load_tile<32, 128, DIMC>(&Bs[buf ^ 1][0][0],
                                         Wt + (size_t)tN * DIMC + k0 + 32);
        }
        AFrag a;                                        // A frag: row=lm, k=8g+e / 16+8g+e
        a.h[0] = *(const v8bf*)&As[buf][wv * 16 + lm][8 * grp];
        a.h[1] = *(const v8bf*)&As[buf][wv * 16 + lm][16 + 8 * grp];
#pragma unroll
        for (int nt = 0; nt < 8; ++nt) {
            AFrag bfr;                                  // B frag: col=lm, k=8g+e / 16+8g+e
            bfr.h[0] = *(const v8bf*)&Bs[buf][nt * 16 + lm][8 * grp];
            bfr.h[1] = *(const v8bf*)&Bs[buf][nt * 16 + lm][16 + 8 * grp];
            acc[nt] = __builtin_amdgcn_wmma_f32_16x16x32_bf16(
                false, a.v, false, bfr.v, (short)0, acc[nt], false, false);
        }
        __syncthreads();                                // readers done before reuse
    }

#pragma unroll
    for (int nt = 0; nt < 8; ++nt)
#pragma unroll
        for (int r = 0; r < 8; ++r) {
            const int row = tM + wv * 16 + r + 8 * grp;
            const int col = tN + nt * 16 + lm;
            const float v = acc[nt][r] + bias[col];
            if (MODE == 2) {
                outf[(size_t)row * DIMC + col] = v;
            } else {
                const int bb = row >> 10, l = row & 1023;
                const int h = col >> 6, d = col & 63;
                const size_t hb = (size_t)bb * NHC + h;
                if (MODE == 0) outb[(hb * LL + l) * DHC + d] = f2bf(v);
                else           outb[(hb * DHC + d) * SS + l] = f2bf(v);
            }
        }
}

// ---------------------------------------------------------------------------
// Flash attention: block = 256 thr (8 waves) on one (b,h) x 128 query rows.
// K/V chunks (32 keys) TDM-staged to LDS, shared by all 8 waves, double-buffered.
// ---------------------------------------------------------------------------
__global__ __launch_bounds__(256) void attn_fa_wmma(
    const bf16* __restrict__ qh, const bf16* __restrict__ kh,
    const bf16* __restrict__ vt, const float* __restrict__ pos,
    const unsigned char* __restrict__ amask, const int* __restrict__ pcausal,
    bf16* __restrict__ o)
{
    __shared__ __attribute__((aligned(16))) bf16 Kt[2][32][64];   // 8 KB (key, dh)
    __shared__ __attribute__((aligned(16))) bf16 Vt[2][64][32];   // 8 KB (dh, key)
    __shared__ __attribute__((aligned(16))) bf16 Ps[8][16][32];   // 8 KB P staging

    const int tid = threadIdx.x;
    const int wv = tid >> 5, lane = tid & 31, grp = lane >> 4, lm = lane & 15;
    const int bh = blockIdx.y, b = bh >> 4, h = bh & 15;
    const int row0 = blockIdx.x * 128 + wv * 16;
    const int causal = *pcausal;
    const float scale = 0.125f;                                   // 1/sqrt(64)

    // per-wave causal bound and uniform per-block bound (barrier-safe)
    int my_send  = causal ? ((row0 + 16 + 31) & ~31) : SS;
    if (my_send > SS) my_send = SS;
    int blk_send = causal ? (int)(blockIdx.x * 128 + 128) : SS;
    if (blk_send > SS) blk_send = SS;

    // Q fragments (16x64 -> two 16x32 A-frags), loaded once
    const bf16* qp = qh + ((size_t)bh * LL + row0 + lm) * DHC;
    AFrag a0, a1;
    a0.h[0] = *(const v8bf*)(qp + 8 * grp);
    a0.h[1] = *(const v8bf*)(qp + 16 + 8 * grp);
    a1.h[0] = *(const v8bf*)(qp + 32 + 8 * grp);
    a1.h[1] = *(const v8bf*)(qp + 48 + 8 * grp);

    v8f acc[4] = {};
    float mrow[8], lrow[8];
#pragma unroll
    for (int r = 0; r < 8; ++r) { mrow[r] = -1e30f; lrow[r] = 0.f; }

    const float* bias_b = pos + (size_t)bh * LL * SS;
    const unsigned char* mk_b = amask + (size_t)b * LL * SS;
    const bf16* kbase = kh + (size_t)bh * SS * DHC;               // (key, dh)
    const bf16* vbase = vt + (size_t)bh * DHC * SS;               // (dh, key)

    if (wv == 0) {                                                // prologue chunk 0
        tdm_load_tile<64, 32, DHC>(&Kt[0][0][0], kbase);
        tdm_load_tile<32, 64, SS>(&Vt[0][0][0], vbase);
    }

    for (int jc = 0; jc < blk_send; jc += 32) {
        const int buf = (jc >> 5) & 1;
        if (wv == 0) tdm_wait0();                                 // chunk jc in LDS
        __syncthreads();
        if (wv == 0 && jc + 32 < blk_send) {                      // overlap next DMA
            tdm_load_tile<64, 32, DHC>(&Kt[buf ^ 1][0][0],
                                       kbase + (size_t)(jc + 32) * DHC);
            tdm_load_tile<32, 64, SS>(&Vt[buf ^ 1][0][0], vbase + jc + 32);
        }

        if (jc < my_send) {
            v8f sc[2];
#pragma unroll
            for (int t = 0; t < 2; ++t) {
                AFrag kb0, kb1;                                   // B frag: col=key(lm)
                kb0.h[0] = *(const v8bf*)&Kt[buf][t * 16 + lm][8 * grp];
                kb0.h[1] = *(const v8bf*)&Kt[buf][t * 16 + lm][16 + 8 * grp];
                kb1.h[0] = *(const v8bf*)&Kt[buf][t * 16 + lm][32 + 8 * grp];
                kb1.h[1] = *(const v8bf*)&Kt[buf][t * 16 + lm][48 + 8 * grp];
                v8f s = {};
                s = __builtin_amdgcn_wmma_f32_16x16x32_bf16(false, a0.v, false, kb0.v,
                                                            (short)0, s, false, false);
                s = __builtin_amdgcn_wmma_f32_16x16x32_bf16(false, a1.v, false, kb1.v,
                                                            (short)0, s, false, false);
                const int col0 = jc + t * 16;
#pragma unroll
                for (int r = 0; r < 8; ++r) {                     // scale+bias+mask
                    const int row = row0 + r + 8 * grp;
                    const int col = col0 + lm;
                    const float bsv = bias_b[(size_t)row * SS + col];
                    bool ok = mk_b[(size_t)row * SS + col] != 0;
                    if (causal && col > row) ok = false;
                    s[r] = ok ? (s[r] * scale + bsv) : -1e30f;
                }
                sc[t] = s;
            }

            if (jc + 32 < my_send)                                // bias stream prefetch
                __builtin_prefetch(bias_b + (size_t)(row0 + lm) * SS + jc + 32, 0, 0);

            float al[8];                                          // online softmax
#pragma unroll
            for (int r = 0; r < 8; ++r) {
                float tm = fmaxf(sc[0][r], sc[1][r]);
                tm = fmaxf(tm, __shfl_xor(tm, 1, 32));
                tm = fmaxf(tm, __shfl_xor(tm, 2, 32));
                tm = fmaxf(tm, __shfl_xor(tm, 4, 32));
                tm = fmaxf(tm, __shfl_xor(tm, 8, 32));
                const float mn = fmaxf(mrow[r], tm);
                const float a  = __expf(mrow[r] - mn);
                mrow[r] = mn;
                const float p0 = __expf(sc[0][r] - mn);
                const float p1 = __expf(sc[1][r] - mn);
                sc[0][r] = p0; sc[1][r] = p1;
                float rs = p0 + p1;
                rs += __shfl_xor(rs, 1, 32);
                rs += __shfl_xor(rs, 2, 32);
                rs += __shfl_xor(rs, 4, 32);
                rs += __shfl_xor(rs, 8, 32);
                lrow[r] = lrow[r] * a + rs;
                al[r] = a;
            }
#pragma unroll
            for (int dt = 0; dt < 4; ++dt)
#pragma unroll
                for (int r = 0; r < 8; ++r) acc[dt][r] *= al[r];

#pragma unroll
            for (int t = 0; t < 2; ++t)                           // P: C->LDS->A layout
#pragma unroll
                for (int r = 0; r < 8; ++r)
                    Ps[wv][r + 8 * grp][t * 16 + lm] = f2bf(sc[t][r]);
            AFrag pf;
            pf.h[0] = *(const v8bf*)&Ps[wv][lm][8 * grp];
            pf.h[1] = *(const v8bf*)&Ps[wv][lm][16 + 8 * grp];

#pragma unroll
            for (int dt = 0; dt < 4; ++dt) {                      // P x V
                AFrag vb;
                vb.h[0] = *(const v8bf*)&Vt[buf][dt * 16 + lm][8 * grp];
                vb.h[1] = *(const v8bf*)&Vt[buf][dt * 16 + lm][16 + 8 * grp];
                acc[dt] = __builtin_amdgcn_wmma_f32_16x16x32_bf16(
                    false, pf.v, false, vb.v, (short)0, acc[dt], false, false);
            }
        }
        __syncthreads();                                          // before buf reuse
    }

    float inv[8];
#pragma unroll
    for (int r = 0; r < 8; ++r) inv[r] = (lrow[r] > 0.f) ? (1.f / lrow[r]) : 0.f;
#pragma unroll
    for (int dt = 0; dt < 4; ++dt)
#pragma unroll
        for (int r = 0; r < 8; ++r) {
            const int row = row0 + r + 8 * grp;
            o[(size_t)(b * LL + row) * DIMC + h * DHC + dt * 16 + lm] =
                f2bf(acc[dt][r] * inv[r]);
        }
}

// ---------------------------------------------------------------------------
extern "C" void kernel_launch(void* const* d_in, const int* in_sizes, int n_in,
                              void* d_out, int out_size, void* d_ws, size_t ws_size,
                              hipStream_t stream) {
    const float* q  = (const float*)d_in[0];
    const float* k  = (const float*)d_in[1];
    const float* v  = (const float*)d_in[2];
    const unsigned char* am = (const unsigned char*)d_in[3];
    const float* pos = (const float*)d_in[4];
    const float* Wq = (const float*)d_in[5],  *bq = (const float*)d_in[6];
    const float* Wk = (const float*)d_in[7],  *bk = (const float*)d_in[8];
    const float* Wv = (const float*)d_in[9],  *bv = (const float*)d_in[10];
    const float* Wp = (const float*)d_in[11], *bp = (const float*)d_in[12];
    const int* isc = (const int*)d_in[13];
    float* out = (float*)d_out;

    const size_t TE = (size_t)BB * LL * DIMC;        // 4 Mi elems per activation
    const size_t WE = (size_t)DIMC * DIMC;           // 1 Mi elems per weight
    bf16* qb  = (bf16*)d_ws;                         // fp32->bf16 activations
    bf16* kb  = qb + TE;
    bf16* vb  = kb + TE;
    bf16* qhh = vb + TE;                             // projected heads
    bf16* khh = qhh + TE;
    bf16* vtr = khh + TE;                            // V transposed (dh, s)
    bf16* wqt = vtr + TE;                            // transposed bf16 weights
    bf16* wkt = wqt + WE;
    bf16* wvt = wkt + WE;
    bf16* wpt = wvt + WE;
    bf16* oh  = qb;                                  // reuse qb region for attn out

    dim3 blk(256);
    cvt_f32_bf16<<<TE / 1024, blk, 0, stream>>>(q, qb);
    cvt_f32_bf16<<<TE / 1024, blk, 0, stream>>>(k, kb);
    cvt_f32_bf16<<<TE / 1024, blk, 0, stream>>>(v, vb);
    dim3 gt(DIMC / 32, DIMC / 32);
    transpose_w_bf16<<<gt, blk, 0, stream>>>(Wq, wqt);
    transpose_w_bf16<<<gt, blk, 0, stream>>>(Wk, wkt);
    transpose_w_bf16<<<gt, blk, 0, stream>>>(Wv, wvt);
    transpose_w_bf16<<<gt, blk, 0, stream>>>(Wp, wpt);

    dim3 gg(DIMC / 128, (BB * LL) / 128);            // (8, 32)
    gemm_tdm_wmma<0><<<gg, blk, 0, stream>>>(qb, wqt, bq, qhh, nullptr);
    gemm_tdm_wmma<0><<<gg, blk, 0, stream>>>(kb, wkt, bk, khh, nullptr);
    gemm_tdm_wmma<1><<<gg, blk, 0, stream>>>(vb, wvt, bv, vtr, nullptr);

    dim3 ga(LL / 128, BB * NHC);                     // (8, 64)
    attn_fa_wmma<<<ga, blk, 0, stream>>>(qhh, khh, vtr, pos, am, isc, oh);

    gemm_tdm_wmma<2><<<gg, blk, 0, stream>>>(oh, wpt, bp, nullptr, out);
}
